// AttentionLayer_84645215469989
// MI455X (gfx1250) — compile-verified, matrix operand order checked by reading, not run on hardware
//
#include <hip/hip_runtime.h>

typedef __attribute__((ext_vector_type(16))) _Float16 v16h;
typedef __attribute__((ext_vector_type(8)))  _Float16 v8h;
typedef __attribute__((ext_vector_type(4)))  _Float16 v4h;
typedef __attribute__((ext_vector_type(8)))  float    v8f;
typedef __attribute__((ext_vector_type(4)))  unsigned u32x4;
typedef __attribute__((ext_vector_type(8)))  int      i32x8;
typedef __attribute__((ext_vector_type(4)))  int      i32x4;

#define B_  8
#define L_  512
#define D_  512
#define H_  8
#define M_  (B_ * L_)    // 4096 rows of X
#define N_  (D_ * H_)    // 4096 projection columns
#define K_  D_           // 512 reduction dim of projections

#define LDT 40           // padded LDS leading dim (halves): row stride 80 B = 5*16 B

// ---------------------------------------------------------------------------
// WMMA fragment loader (CDNA5 wave32 layouts). Works for both operands:
//  A (16x32, MxK) from tile[row][k]   : row = lane&15
//  B (32x16, KxN) from tileT[col][k]  : col = lane&15 (tile staged transposed)
// Per lane the fragment is two contiguous 8-half runs -> two ds_load_b128.
// ---------------------------------------------------------------------------
__device__ inline v16h load_frag(const _Float16* tile, int ld, int lane) {
  const int row = lane & 15;
  const int kb  = (lane >> 4) << 3;
  const _Float16* p = tile + row * ld + kb;
  const v8h lo = *(const v8h*)(p);          // K = kb .. kb+7
  const v8h hi = *(const v8h*)(p + 16);     // K = 16+kb .. 16+kb+7
  return __builtin_shufflevector(lo, hi, 0,1,2,3,4,5,6,7,8,9,10,11,12,13,14,15);
}

// ---------------------------------------------------------------------------
// Tensor Data Mover: 2D tile global->LDS per ISA cdna5 D# (groups 0/1; 2/3 zero).
// data_size = 2 bytes. lds_off = wave-relative LDS byte address (= flat addr
// low 32 bits per ISA LDS aperture rule). Issue from ONE wave only.
// ---------------------------------------------------------------------------
__device__ inline void tdm_load_2d_f16(unsigned lds_off, const void* gaddr,
                                       unsigned tdim0, unsigned tdim1,
                                       unsigned tile0, unsigned tile1,
                                       unsigned stride0 /* elements */) {
  const unsigned long long ga = (unsigned long long)gaddr;
  u32x4 g0;
  g0.x = 1u;                                         // count=1, user mode
  g0.y = lds_off;                                    // lds_addr
  g0.z = (unsigned)(ga & 0xFFFFFFFFull);             // global_addr[31:0]
  g0.w = (unsigned)((ga >> 32) & 0x1FFFFFFull)       // global_addr[56:32]
         | 0x80000000u;                              // type=2 ("image")
  i32x8 g1;
  g1[0] = (int)(1u << 16);                                           // data_size=2B
  g1[1] = (int)((tdim0 & 0xFFFFu) << 16);                            // tensor_dim0 lo
  g1[2] = (int)(((tdim0 >> 16) & 0xFFFFu) | ((tdim1 & 0xFFFFu) << 16));
  g1[3] = (int)(((tdim1 >> 16) & 0xFFFFu) | ((tile0 & 0xFFFFu) << 16));
  g1[4] = (int)(tile1 & 0xFFFFu);                                    // tile_dim1 (dim2=0)
  g1[5] = (int)stride0;                                              // dim0_stride lo
  g1[6] = 0;                                                         // stride hi / dim1_stride lo
  g1[7] = 0;
  const i32x4 z4 = {0, 0, 0, 0};
#if __clang_major__ >= 23
  const i32x8 z8 = {0, 0, 0, 0, 0, 0, 0, 0};
  __builtin_amdgcn_tensor_load_to_lds(g0, g1, z4, z4, z8, 0);
#else
  __builtin_amdgcn_tensor_load_to_lds(g0, g1, z4, z4, 0);
#endif
}

// ---------------------------------------------------------------------------
// Kernel A: Out(f16) = X(f32) @ W(f32) + bias. M=4096, N=4096, K=512.
// 64x64 tile per block, 128 threads = 4 waves. TS=true stores V transposed:
// VhT[b][e][l*8+g] so the PV GEMM's B operand becomes a plain row-major tile
// (TDM-movable, no LDS scatter).
// ---------------------------------------------------------------------------
template <bool TS>
__global__ __launch_bounds__(128)
void proj_gemm_f16(const float* __restrict__ X, const float* __restrict__ W,
                   const float* __restrict__ bias, _Float16* __restrict__ Out) {
  __shared__ alignas(16) _Float16 As [64][LDT];   // A tile  64(M) x 32(K)
  __shared__ alignas(16) _Float16 BsT[64][LDT];   // B tile transposed: 64(N) x 32(K)
  const int tid  = threadIdx.x;
  const int lane = tid & 31;
  const int wave = tid >> 5;
  const int rowBase = blockIdx.y * 64;
  const int colBase = blockIdx.x * 64;

  v8f acc[4];
#pragma unroll
  for (int n = 0; n < 4; ++n) acc[n] = (v8f){0.f,0.f,0.f,0.f,0.f,0.f,0.f,0.f};

  for (int k0 = 0; k0 < K_; k0 += 32) {
    // ---- stage A 64x32: float4 global loads, packed b64 LDS stores ----
#pragma unroll
    for (int i = 0; i < 4; ++i) {
      const int idx = tid * 4 + i * 512;          // 0..2047
      const int r = idx >> 5, c = idx & 31;       // c multiple of 4
      const float4 x = *(const float4*)&X[(size_t)(rowBase + r) * K_ + k0 + c];
      const v4h hx = { (_Float16)x.x, (_Float16)x.y, (_Float16)x.z, (_Float16)x.w };
      *(v4h*)&As[r][c] = hx;
    }
    // ---- stage B 32x64 transposed: float4 global loads, b16 LDS scatter ----
#pragma unroll
    for (int i = 0; i < 4; ++i) {
      const int idx = tid * 4 + i * 512;
      const int r = idx >> 6, c = idx & 63;       // c multiple of 4
      const float4 wv = *(const float4*)&W[(size_t)(k0 + r) * N_ + colBase + c];
      BsT[c + 0][r] = (_Float16)wv.x;
      BsT[c + 1][r] = (_Float16)wv.y;
      BsT[c + 2][r] = (_Float16)wv.z;
      BsT[c + 3][r] = (_Float16)wv.w;
    }
    if (k0 + 32 < K_) {                           // hint next k-tile into cache
      __builtin_prefetch(&X[(size_t)(rowBase + (tid >> 2)) * K_ + k0 + 32], 0, 1);
      __builtin_prefetch(&W[(size_t)(k0 + 32 + (tid >> 2)) * N_ + colBase], 0, 1);
    }
    __syncthreads();
    const v16h a = load_frag(&As[wave * 16][0], LDT, lane);
#pragma unroll
    for (int n = 0; n < 4; ++n) {
      const v16h bfrag = load_frag(&BsT[n * 16][0], LDT, lane);
      acc[n] = __builtin_amdgcn_wmma_f32_16x16x32_f16(
          false, a, false, bfrag, (short)0, acc[n], false, false);
    }
    __syncthreads();
  }

  // epilogue: D mapping reg r -> row = r + (lane>=16?8:0), col = lane&15
  const int cl = lane & 15;
  const int rh = (lane >> 4) << 3;
#pragma unroll
  for (int n = 0; n < 4; ++n) {
    const int col  = colBase + n * 16 + cl;
    const float bv = bias[col];
#pragma unroll
    for (int r = 0; r < 8; ++r) {
      const int row = rowBase + wave * 16 + rh + r;     // = b*L + l
      const _Float16 val = (_Float16)(acc[n][r] + bv);
      if (TS) {
        const int e = col & (D_ - 1), g = col >> 9;     // col = g*512 + e
        const int bb = row >> 9, ll = row & (L_ - 1);
        Out[(size_t)(bb * D_ + e) * (L_ * H_) + ll * H_ + g] = val;
      } else {
        Out[(size_t)row * N_ + col] = val;
      }
    }
  }
}

// ---------------------------------------------------------------------------
// Kernel B: per position (b,l): s[h,g] = (q_h . k_g)/sqrt(D); softmax over g.
// Pt[b][row 0..7][l*8+g] = p; row 8 = (L-1)/H (uniform term folded into the
// GEMM); rows 9..15 = 0 (M padding to 16).
// ---------------------------------------------------------------------------
__global__ __launch_bounds__(64)
void scores_softmax(const _Float16* __restrict__ Qh, const _Float16* __restrict__ Kh,
                    _Float16* __restrict__ Pt) {
  __shared__ float s[64];
  const int pos = blockIdx.x;              // b*L + l
  const int b = pos / L_, l = pos % L_;
  const int tid = threadIdx.x;
  const int h = tid >> 3, g = tid & 7;

  const _Float16* qrow = Qh + (size_t)pos * N_ + h * D_;   // 16B aligned
  const _Float16* krow = Kh + (size_t)pos * N_ + g * D_;
  float dot = 0.f;
#pragma unroll 4
  for (int e = 0; e < D_; e += 8) {
    const v8h qv = *(const v8h*)(qrow + e);
    const v8h kv = *(const v8h*)(krow + e);
#pragma unroll
    for (int j = 0; j < 8; ++j) dot += (float)qv[j] * (float)kv[j];
  }
  s[tid] = dot * 0.044194173824159216f;    // 1/sqrt(512)
  __syncthreads();

  _Float16* pbase = Pt + (size_t)b * 16 * (L_ * H_);
  if (tid < 8) {
    float mx = s[tid * 8];
    for (int j = 1; j < 8; ++j) mx = fmaxf(mx, s[tid * 8 + j]);
    float ex[8], sum = 0.f;
    for (int j = 0; j < 8; ++j) { ex[j] = __expf(s[tid * 8 + j] - mx); sum += ex[j]; }
    const float inv = 1.f / sum;
    for (int j = 0; j < 8; ++j)
      pbase[(size_t)tid * (L_ * H_) + l * 8 + j] = (_Float16)(ex[j] * inv);
  } else if (tid < 16) {
    const _Float16 val = (tid == 8) ? (_Float16)(511.0f / 8.0f) : (_Float16)0.f;
    for (int j = 0; j < 8; ++j)
      pbase[(size_t)tid * (L_ * H_) + l * 8 + j] = val;
  }
}

// ---------------------------------------------------------------------------
// Kernel C: per batch b: D(16x512) = Pt[b](16x4096) @ V[b](4096x512).
// Both tiles are plain row-major rectangles, DMA'd by the Tensor Data Mover
// into double-buffered LDS: wave0 waits tensorcnt for tile i, then launches
// the DMAs for tile i+1 so they run UNDER the WMMAs of tile i.
// out[b,h,e] = D[h,e] + D[8,e] (row 8 carries the folded uniform term).
// ---------------------------------------------------------------------------
__global__ __launch_bounds__(128)
void pv_gemm(const _Float16* __restrict__ Pt, const _Float16* __restrict__ VhT,
             float* __restrict__ Out) {
  __shared__ alignas(16) _Float16 As2 [2][16][32];  // packed: TDM writes contiguously
  __shared__ alignas(16) _Float16 BsT2[2][64][32];
  __shared__ float Ct[4][16][16];
  const int tid = threadIdx.x, lane = tid & 31, wave = tid >> 5;
  const int b = blockIdx.y;
  const int colBase = blockIdx.x * 64;            // output column block (e)
  const int KK = L_ * H_;                         // 4096
  const _Float16* A  = Pt  + (size_t)b * 16 * KK;          // 16 x 4096 row-major
  const _Float16* Bt = VhT + (size_t)b * D_ * KK            // 512(e) x 4096(k) row-major
                       + (size_t)colBase * KK;

  unsigned ldsA[2], ldsB[2];
#pragma unroll
  for (int p = 0; p < 2; ++p) {
    ldsA[p] = (unsigned)(unsigned long long)&As2[p][0][0];
    ldsB[p] = (unsigned)(unsigned long long)&BsT2[p][0][0];
  }

  v8f acc = (v8f){0.f,0.f,0.f,0.f,0.f,0.f,0.f,0.f};

  if (wave == 0) {                                // prologue: fill buffer 0
    tdm_load_2d_f16(ldsA[0], A,  KK, 16, 32, 16, KK);
    tdm_load_2d_f16(ldsB[0], Bt, KK, D_, 32, 64, KK);
  }

  int buf = 0;
  for (int k0 = 0; k0 < KK; k0 += 32, buf ^= 1) {
    if (wave == 0) __builtin_amdgcn_s_wait_tensorcnt(0);   // tile k0 landed
    __syncthreads();                                        // publish to all waves
    const int nk = k0 + 32;
    if (wave == 0 && nk < KK) {                  // DMA next tile under the math
      tdm_load_2d_f16(ldsA[buf ^ 1], A + nk,  KK, 16, 32, 16, KK);
      tdm_load_2d_f16(ldsB[buf ^ 1], Bt + nk, KK, D_, 32, 64, KK);
    }
    const v16h a  = load_frag(&As2[buf][0][0], 32, lane);
    const v16h bf = load_frag(&BsT2[buf][wave * 16][0], 32, lane);
    acc = __builtin_amdgcn_wmma_f32_16x16x32_f16(
        false, a, false, bf, (short)0, acc, false, false);
    __syncthreads();                             // all reads of buf done
  }

  {
    const int cl = lane & 15, rh = (lane >> 4) << 3;
#pragma unroll
    for (int r = 0; r < 8; ++r) Ct[wave][rh + r][cl] = acc[r];
  }
  __syncthreads();

  for (int idx = tid; idx < 512; idx += 128) {    // 4 waves * 8 heads * 16 cols
    const int w = idx >> 7, rem = idx & 127, h = rem >> 4, cl = rem & 15;
    const int col = colBase + w * 16 + cl;
    Out[(size_t)b * (H_ * D_) + h * D_ + col] = Ct[w][h][cl] + Ct[w][8][cl];
  }
}

// ---------------------------------------------------------------------------
extern "C" void kernel_launch(void* const* d_in, const int* in_sizes, int n_in,
                              void* d_out, int out_size, void* d_ws, size_t ws_size,
                              hipStream_t stream) {
  (void)in_sizes; (void)n_in; (void)out_size; (void)ws_size;
  const float* q  = (const float*)d_in[0];
  const float* k  = (const float*)d_in[1];
  const float* v  = (const float*)d_in[2];
  const float* Wq = (const float*)d_in[3];
  const float* bq = (const float*)d_in[4];
  const float* Wk = (const float*)d_in[5];
  const float* bk = (const float*)d_in[6];
  const float* Wv = (const float*)d_in[7];
  const float* bv = (const float*)d_in[8];
  float* out = (float*)d_out;

  // Workspace (f16): Qh | Kh | VhT (32 MB each) | Pt (1 MB) = ~97 MB < 192 MB L2.
  const size_t szProj = (size_t)M_ * N_ * sizeof(_Float16);
  _Float16* Qh  = (_Float16*)d_ws;
  _Float16* Kh  = (_Float16*)((char*)d_ws + szProj);
  _Float16* VhT = (_Float16*)((char*)d_ws + 2 * szProj);
  _Float16* Pt  = (_Float16*)((char*)d_ws + 3 * szProj);

  const dim3 gA(N_ / 64, M_ / 64);   // 64 x 64 blocks
  proj_gemm_f16<false><<<gA, 128, 0, stream>>>(q, Wq, bq, Qh);
  proj_gemm_f16<false><<<gA, 128, 0, stream>>>(k, Wk, bk, Kh);
  proj_gemm_f16<true ><<<gA, 128, 0, stream>>>(v, Wv, bv, VhT);

  scores_softmax<<<M_, 64, 0, stream>>>(Qh, Kh, Pt);

  pv_gemm<<<dim3(D_ / 64, B_), 128, 0, stream>>>(Pt, VhT, out);
}
